// PolicyGradient_pt_36558761624242
// MI455X (gfx1250) — matrix operand order repaired and checked by reading, not compiled, not run on hardware
//
#include <hip/hip_runtime.h>
#include <hip/hip_bf16.h>

typedef __attribute__((ext_vector_type(16))) _Float16 v16h;
typedef __attribute__((ext_vector_type(8)))  float    v8f;

#define N_ROWS   131072
#define IN_DIM   256
#define OUT_DIM  64
#define H1_DIM   64
#define H2_DIM   32
#define W1_COLS  (IN_DIM + OUT_DIM)   // 320
#define BN_EPS   1e-5f

#define STAT_BLOCKS 256
#define ROWS_PER_STAT_BLOCK (N_ROWS / STAT_BLOCKS)  // 512

// ---------------------------------------------------------------------------
// Kernel 1: per-block partial column sums / sums-of-squares (coalesced,
// deterministic: fixed partial layout, no float atomics).
// ---------------------------------------------------------------------------
__global__ void __launch_bounds__(256)
k_stats_partial(const float* __restrict__ x,
                float* __restrict__ psum, float* __restrict__ psq) {
    const int c = threadIdx.x;          // column 0..255
    const int b = blockIdx.x;           // row-chunk 0..255
    const float* xp = x + (size_t)b * ROWS_PER_STAT_BLOCK * IN_DIM + c;
    float s = 0.0f, q = 0.0f;
    #pragma unroll 4
    for (int r = 0; r < ROWS_PER_STAT_BLOCK; ++r) {
        float v = xp[(size_t)r * IN_DIM];
        s += v;
        q = fmaf(v, v, q);
    }
    psum[b * IN_DIM + c] = s;
    psq [b * IN_DIM + c] = q;
}

// ---------------------------------------------------------------------------
// Kernel 2: finalize BN affine:  xn = x*scale[c] + shift[c]
// ---------------------------------------------------------------------------
__global__ void __launch_bounds__(256)
k_stats_finalize(const float* __restrict__ psum, const float* __restrict__ psq,
                 const float* __restrict__ bnw, const float* __restrict__ bnb,
                 float* __restrict__ scale, float* __restrict__ shift) {
    const int c = threadIdx.x;
    float s = 0.0f, q = 0.0f;
    for (int b = 0; b < STAT_BLOCKS; ++b) {
        s += psum[b * IN_DIM + c];
        q += psq [b * IN_DIM + c];
    }
    const float inv_n = 1.0f / (float)N_ROWS;
    const float mean  = s * inv_n;
    const float var   = q * inv_n - mean * mean;     // biased, matches reference
    const float sc    = rsqrtf(var + BN_EPS) * bnw[c];
    scale[c] = sc;
    shift[c] = fmaf(-mean, sc, bnb[c]);
}

// ---------------------------------------------------------------------------
// Kernel 3: U = BN(x) @ W1x^T + b1   via v_wmma_f32_16x16x32_f16.
// One block = 8 waves, each wave owns a 16-row x 64-col strip of U.
// W1x^T is pre-swizzled in LDS into per-lane B-fragment order so each
// fragment is a single contiguous 32B LDS read.
// ---------------------------------------------------------------------------
#define GEMM_WAVES 8
#define GEMM_BLOCK_ROWS (GEMM_WAVES * 16)   // 128 rows / block -> 1024 blocks

__global__ void __launch_bounds__(256)
k_gemm_u(const float* __restrict__ x, const float* __restrict__ W1,
         const float* __restrict__ b1, const float* __restrict__ scale,
         const float* __restrict__ shift, float* __restrict__ U) {
    // B fragments: [kc:8][ct:4][lane:32][16 halves] = 16384 halves = 32KB
    __shared__ __align__(32) _Float16 Bswz[8 * 4 * 32 * 16];
    __shared__ float s_scale[IN_DIM];
    __shared__ float s_shift[IN_DIM];
    __shared__ float s_b1[H1_DIM];

    const int tid = threadIdx.x;

    // One-time swizzled conversion of W1x^T (f32 -> f16) into LDS.
    // B[k][n] = W1[n][k]; lane layout: halves 0..7 hold K=kb+sel*8+{0..7},
    // halves 8..15 hold K=kb+16+sel*8+{0..7} (n=lane&15), 2 halves per VGPR.
    for (int f = tid; f < 8 * 4 * 32 * 16; f += 256) {
        const int j    = f & 15;
        const int lane = (f >> 4) & 31;
        const int ct   = (f >> 9) & 3;
        const int kc   = f >> 11;
        const int n    = ct * 16 + (lane & 15);
        const int sel  = lane >> 4;
        const int k    = kc * 32 + (j >> 3) * 16 + sel * 8 + (j & 7);
        Bswz[f] = (_Float16)W1[n * W1_COLS + k];
    }
    if (tid < IN_DIM) { s_scale[tid] = scale[tid]; s_shift[tid] = shift[tid]; }
    if (tid < H1_DIM) { s_b1[tid] = b1[tid]; }
    __syncthreads();

    const int wave    = tid >> 5;
    const int lane    = tid & 31;
    const int m       = lane & 15;
    const int sel     = lane >> 4;
    const int rowBase = blockIdx.x * GEMM_BLOCK_ROWS + wave * 16;
    const float* xrow = x + (size_t)(rowBase + m) * IN_DIM;

    v8f acc[4];
    #pragma unroll
    for (int ct = 0; ct < 4; ++ct)
        #pragma unroll
        for (int r = 0; r < 8; ++r) acc[ct][r] = 0.0f;

    #pragma unroll 2
    for (int kc = 0; kc < 8; ++kc) {
        // A fragment (16-bit A 16x32 layout): lane m holds
        //   halves 0..7  -> K = kc*32 + sel*8 + {0..7}
        //   halves 8..15 -> K = kc*32 + 16 + sel*8 + {0..7}
        v16h afrag;
        #pragma unroll
        for (int p = 0; p < 2; ++p) {
            const int kb = kc * 32 + p * 16 + sel * 8;
            const float4 lo = *(const float4*)(xrow + kb);
            const float4 hi = *(const float4*)(xrow + kb + 4);
            const float xv[8] = {lo.x, lo.y, lo.z, lo.w, hi.x, hi.y, hi.z, hi.w};
            #pragma unroll
            for (int t = 0; t < 8; ++t) {
                const float xn = fmaf(xv[t], s_scale[kb + t], s_shift[kb + t]);
                afrag[p * 8 + t] = (_Float16)xn;
            }
        }
        #pragma unroll
        for (int ct = 0; ct < 4; ++ct) {
            const v16h bfrag =
                *(const v16h*)&Bswz[(((kc * 4 + ct) * 32 + lane) << 4)];
            acc[ct] = __builtin_amdgcn_wmma_f32_16x16x32_f16(
                false, afrag, false, bfrag, (short)0, acc[ct], false, false);
        }
    }

    // D layout: VGPR r, lanes 0-15 -> M=r, lanes 16-31 -> M=r+8; N = lane&15.
    #pragma unroll
    for (int ct = 0; ct < 4; ++ct) {
        const float bb = s_b1[ct * 16 + m];
        #pragma unroll
        for (int r = 0; r < 8; ++r) {
            const int orow = rowBase + sel * 8 + r;
            U[(size_t)orow * OUT_DIM + ct * 16 + m] = acc[ct][r] + bb;
        }
    }
}

// ---------------------------------------------------------------------------
// Kernel 4: the serial recurrence. Single wave32; weights LDS-resident with
// row padding (65 / 33) so lane-indexed rows hit distinct banks. Barriers in
// a single-wave workgroup are S_NOPs; only dscnt fences remain.
// ---------------------------------------------------------------------------
__global__ void __launch_bounds__(32)
k_scan(const float* __restrict__ U, const float* __restrict__ W1,
       const float* __restrict__ W2, const float* __restrict__ b2,
       const float* __restrict__ W3, const float* __restrict__ b3,
       float* __restrict__ out) {
    __shared__ float W1h_s[64][65];
    __shared__ float W2_s[32][65];
    __shared__ float W3_s[64][33];
    __shared__ float b2_s[32];
    __shared__ float b3_s[64];
    __shared__ float h_s[64];
    __shared__ float a1_s[64];
    __shared__ float a2_s[32];

    const int l = threadIdx.x;   // 0..31

    for (int e = l; e < 64 * 64; e += 32) {
        const int j = e >> 6, k = e & 63;
        W1h_s[j][k] = W1[j * W1_COLS + IN_DIM + k];
    }
    for (int e = l; e < 32 * 64; e += 32) {
        const int j = e >> 6, k = e & 63;
        W2_s[j][k] = W2[j * 64 + k];
    }
    for (int e = l; e < 64 * 32; e += 32) {
        const int j = e >> 5, k = e & 31;
        W3_s[j][k] = W3[j * 32 + k];
    }
    b2_s[l] = b2[l];
    b3_s[l] = b3[l];  b3_s[l + 32] = b3[l + 32];
    h_s[l]  = 0.0f;   h_s[l + 32]  = 0.0f;
    __syncthreads();

    for (int i = 0; i < N_ROWS; ++i) {
        const float* u = U + (size_t)i * OUT_DIM;
        __builtin_prefetch(U + (size_t)(i + 16) * OUT_DIM, 0, 1);  // global_prefetch_b8

        // a1 = relu(u + W1h @ h)   (u already contains W1x@xn + b1)
        float s0 = u[l];
        float s1 = u[l + 32];
        #pragma unroll 4
        for (int k = 0; k < 64; ++k) {
            const float hk = h_s[k];                 // LDS broadcast
            s0 = fmaf(W1h_s[l][k],      hk, s0);     // bank (l+k)%64: conflict-free
            s1 = fmaf(W1h_s[l + 32][k], hk, s1);
        }
        s0 = fmaxf(s0, 0.0f);
        s1 = fmaxf(s1, 0.0f);
        a1_s[l] = s0;  a1_s[l + 32] = s1;
        __syncthreads();

        // a2 = relu(W2 @ a1 + b2)
        float t0 = b2_s[l];
        #pragma unroll 4
        for (int k = 0; k < 64; ++k)
            t0 = fmaf(W2_s[l][k], a1_s[k], t0);
        t0 = fmaxf(t0, 0.0f);
        a2_s[l] = t0;
        __syncthreads();

        // o = W3 @ a2 + b3 ;  h <- o
        float o0 = b3_s[l];
        float o1 = b3_s[l + 32];
        #pragma unroll 4
        for (int k = 0; k < 32; ++k) {
            const float ak = a2_s[k];
            o0 = fmaf(W3_s[l][k],      ak, o0);
            o1 = fmaf(W3_s[l + 32][k], ak, o1);
        }
        out[(size_t)i * OUT_DIM + l]      = o0;
        out[(size_t)i * OUT_DIM + l + 32] = o1;
        h_s[l] = o0;  h_s[l + 32] = o1;
        __syncthreads();
    }
}

// ---------------------------------------------------------------------------
// Host launcher
// ---------------------------------------------------------------------------
extern "C" void kernel_launch(void* const* d_in, const int* in_sizes, int n_in,
                              void* d_out, int out_size, void* d_ws, size_t ws_size,
                              hipStream_t stream) {
    (void)in_sizes; (void)n_in; (void)out_size; (void)ws_size;

    const float* x   = (const float*)d_in[0];
    const float* bnw = (const float*)d_in[1];
    const float* bnb = (const float*)d_in[2];
    const float* W1  = (const float*)d_in[3];
    const float* b1  = (const float*)d_in[4];
    const float* W2  = (const float*)d_in[5];
    const float* b2  = (const float*)d_in[6];
    const float* W3  = (const float*)d_in[7];
    const float* b3  = (const float*)d_in[8];
    float* out = (float*)d_out;

    // Workspace layout (floats): partials, BN affine, U precompute.
    float* ws    = (float*)d_ws;
    float* psum  = ws;                               // 256*256
    float* psq   = psum  + STAT_BLOCKS * IN_DIM;     // 256*256
    float* scale = psq   + STAT_BLOCKS * IN_DIM;     // 256
    float* shift = scale + IN_DIM;                   // 256
    float* U     = shift + IN_DIM;                   // N*64

    k_stats_partial <<<STAT_BLOCKS, 256, 0, stream>>>(x, psum, psq);
    k_stats_finalize<<<1,           256, 0, stream>>>(psum, psq, bnw, bnb, scale, shift);
    k_gemm_u        <<<N_ROWS / GEMM_BLOCK_ROWS, 256, 0, stream>>>(x, W1, b1, scale, shift, U);
    k_scan          <<<1,            32, 0, stream>>>(U, W1, W2, b2, W3, b3, out);
}
